// GNN_14645838480120
// MI455X (gfx1250) — compile-verified, matrix operand order checked by reading, not run on hardware
//
#include <hip/hip_runtime.h>
#include <stdint.h>

// ---------------- problem constants ----------------
#define N_NODES   100000
#define N_EDGES   640000
#define D_IN      128
#define HIDDEN    256
#define OUT_DIM   256
#define NUM_GRAPHS 256

// ---------------- types ----------------
typedef __attribute__((ext_vector_type(16))) __bf16 v16bf;
typedef __attribute__((ext_vector_type(8)))  float  v8f;

union FragBF {
    unsigned int u[8];
    v16bf v;
};

__device__ __forceinline__ unsigned short f32_to_bf16(float f) {
    unsigned int u = __float_as_uint(f);
    u += 0x7FFFu + ((u >> 16) & 1u);   // round-to-nearest-even
    return (unsigned short)(u >> 16);
}
__device__ __forceinline__ float bf16u_to_f32(unsigned int us) {
    return __uint_as_float(us << 16);
}

// ---------------- weight packer: f32 [K,256] -> bf16 WMMA-B fragment order ---
// wp[((kstep*16 + coltile)*32 + lane)*16 + i]  matches the per-lane load order
// in the GEMM (two uint4 = 16 bf16 per lane per k-step).
__global__ void gnn_pack_weights(const float* __restrict__ Wl,
                                 const float* __restrict__ Wr,
                                 int K1, unsigned short* __restrict__ wp,
                                 int total) {
    int idx = blockIdx.x * blockDim.x + threadIdx.x;
    if (idx >= total) return;
    int i    = idx & 15;
    int lane = (idx >> 4) & 31;
    int ct   = (idx >> 9) & 15;
    int s    = idx >> 13;
    int h  = lane >> 4;
    int nl = lane & 15;
    int v  = i >> 1;
    int p  = i & 1;
    // 16-bit matrix K-pattern (ISA 7.12.2): VGPR 0..3 -> K = 8h+2v+p, VGPR 4..7 -> 16+8h+2(v-4)+p
    int klocal = (v < 4) ? (8 * h + 2 * v + p) : (16 + 8 * h + 2 * (v - 4) + p);
    int k = s * 32 + klocal;
    int n = ct * 16 + nl;
    float w = (k < K1) ? Wl[(size_t)k * 256 + n] : Wr[(size_t)(k - K1) * 256 + n];
    wp[idx] = f32_to_bf16(w);
}

// ---------------- degree count ----------------
__global__ void gnn_count_edges(const long long* __restrict__ dst,
                                float* __restrict__ cnt) {
    int e = blockIdx.x * 256 + threadIdx.x;
    if (e < N_EDGES) atomicAdd(&cnt[dst[e]], 1.0f);
}

// ---------------- layer-1 edge scatter: agg1[dst] += x[src], d=128 f32 -------
__global__ void gnn_edge_agg_f32(const float* __restrict__ x,
                                 const long long* __restrict__ src,
                                 const long long* __restrict__ dst,
                                 float* __restrict__ agg) {
    int e    = blockIdx.x * 8 + (threadIdx.x >> 5);   // one wave32 per edge
    int lane = threadIdx.x & 31;
    long long s = src[e];
    long long d = dst[e];
    const float4* xs = (const float4*)(x + (size_t)s * D_IN);
    float4 v = xs[lane];                              // 128 f32 = 32 lanes x float4
    float* ap = agg + (size_t)d * D_IN + lane * 4;
    atomicAdd(ap + 0, v.x);
    atomicAdd(ap + 1, v.y);
    atomicAdd(ap + 2, v.z);
    atomicAdd(ap + 3, v.w);
}

// ---------------- layer-2 edge scatter: agg2[dst] += h[src], d=256 bf16 ------
__global__ void gnn_edge_agg_bf16(const unsigned short* __restrict__ h,
                                  const long long* __restrict__ src,
                                  const long long* __restrict__ dst,
                                  float* __restrict__ agg) {
    int e    = blockIdx.x * 8 + (threadIdx.x >> 5);
    int lane = threadIdx.x & 31;
    long long s = src[e];
    long long d = dst[e];
    const uint4* hs = (const uint4*)(h + (size_t)s * HIDDEN); // 256 bf16 = 32 x 16B
    uint4 v = hs[lane];
    float* ap = agg + (size_t)d * HIDDEN + lane * 8;
    unsigned int uu[4] = {v.x, v.y, v.z, v.w};
#pragma unroll
    for (int j = 0; j < 4; ++j) {
        atomicAdd(ap + 2 * j + 0, bf16u_to_f32(uu[j] & 0xFFFFu));
        atomicAdd(ap + 2 * j + 1, bf16u_to_f32(uu[j] >> 16));
    }
}

// ---------------- fused SAGE GEMM -------------------------------------------
// C[16 x 256] = [ agg/max(cnt,1) | A2 ] @ Wpack + bias, per block of 16 nodes.
// 512 threads = 16 waves, wave w owns column tile w. A tile staged to LDS in
// bf16 with +8-ushort row pad to avoid bank conflicts on ds_load_b128.
// POOL=false: relu -> bf16 store to hout.  POOL=true: atomicAdd into
// pout[batch[node]*256 + col] (fused global_add_pool).
template <int K1, int K2, bool A2BF, bool POOL>
__global__ __launch_bounds__(512) void gnn_sage_gemm(
    const float* __restrict__ agg, const float* __restrict__ cnt,
    const float* __restrict__ a2f, const unsigned short* __restrict__ a2b,
    const unsigned short* __restrict__ wpack, const float* __restrict__ bias,
    unsigned short* __restrict__ hout, float* __restrict__ pout,
    const long long* __restrict__ batch) {
    constexpr int K  = K1 + K2;
    constexpr int KS = K / 32;                 // k-steps of 32
    constexpr int RS_US = K + 8;               // padded row stride in ushorts
    constexpr int RS_V4 = K / 8 + 1;           // row stride in uint4

    __shared__ unsigned int ldsA32[16 * (K / 2 + 4)];
    unsigned short* ldsS = (unsigned short*)ldsA32;

    const int tid       = threadIdx.x;
    const int node_base = blockIdx.x * 16;

    // ---- stage A tile (16 rows x K) into LDS as bf16 ----
    constexpr int PER = (16 * K) / 512;
#pragma unroll
    for (int j = 0; j < PER; ++j) {
        int e    = tid * PER + j;
        int row  = e / K;
        int k    = e % K;
        int node = node_base + row;
        float v;
        if (k < K1) {
            float c = cnt[node];
            float s = 1.0f / fmaxf(c, 1.0f);
            v = agg[(size_t)node * K1 + k] * s;
        } else {
            int k2 = k - K1;
            if (A2BF) v = bf16u_to_f32((unsigned int)a2b[(size_t)node * K2 + k2]);
            else      v = a2f[(size_t)node * K2 + k2];
        }
        ldsS[row * RS_US + k] = f32_to_bf16(v);
    }
    __syncthreads();

    const int wave = tid >> 5;        // column tile 0..15
    const int lane = tid & 31;
    const int half = lane >> 4;
    const int nl   = lane & 15;

    const uint4* Av = (const uint4*)ldsA32;
    const uint4* Bv = (const uint4*)wpack;

    v8f acc = {};
    FragBF a, b;
#pragma unroll
    for (int s = 0; s < KS; ++s) {
        int ai = nl * RS_V4 + s * 4 + half;
        *(uint4*)&a.u[0] = Av[ai];
        *(uint4*)&a.u[4] = Av[ai + 2];
        int bi = ((s * 16 + wave) * 32 + lane) * 2;
        *(uint4*)&b.u[0] = Bv[bi];
        *(uint4*)&b.u[4] = Bv[bi + 1];
        acc = __builtin_amdgcn_wmma_f32_16x16x32_bf16(
            false, a.v, false, b.v, (short)0, acc, false, false);
    }

    // ---- epilogue (C/D layout: lane half -> N=nl, VGPR r -> M = r + 8*half) --
    const int col = wave * 16 + nl;
    const float bb = bias[col];
#pragma unroll
    for (int r = 0; r < 8; ++r) {
        int node = node_base + r + 8 * half;
        float v = acc[r] + bb;
        if (POOL) {
            int g = (int)batch[node];
            atomicAdd(&pout[(size_t)g * OUT_DIM + col], v);
        } else {
            v = fmaxf(v, 0.0f);
            hout[(size_t)node * HIDDEN + col] = f32_to_bf16(v);
        }
    }
}

// ---------------- launch ----------------
extern "C" void kernel_launch(void* const* d_in, const int* in_sizes, int n_in,
                              void* d_out, int out_size, void* d_ws, size_t ws_size,
                              hipStream_t stream) {
    (void)in_sizes; (void)n_in; (void)ws_size;
    const float*     x     = (const float*)d_in[0];
    const long long* ei    = (const long long*)d_in[1];
    const long long* batch = (const long long*)d_in[2];
    const float*     W1l   = (const float*)d_in[3];
    const float*     b1    = (const float*)d_in[4];
    const float*     W1r   = (const float*)d_in[5];
    const float*     W2l   = (const float*)d_in[6];
    const float*     b2    = (const float*)d_in[7];
    const float*     W2r   = (const float*)d_in[8];
    float* out = (float*)d_out;

    const long long* src = ei;
    const long long* dst = ei + N_EDGES;

    // workspace layout (agg1 aliases the front of the agg2 buffer; agg2 is
    // re-zeroed after GEMM1 in stream order)
    char*  ws  = (char*)d_ws;
    size_t off = 0;
    auto take = [&](size_t bytes) { char* p = ws + off; off += (bytes + 255) & ~(size_t)255; return p; };
    float*          buf  = (float*)take((size_t)N_NODES * HIDDEN * 4); // agg1 (N*128 f32) then agg2 (N*256 f32)
    float*          cnt  = (float*)take((size_t)N_NODES * 4);
    unsigned short* hbf  = (unsigned short*)take((size_t)N_NODES * HIDDEN * 2);
    unsigned short* wp1  = (unsigned short*)take((size_t)65536 * 2);
    unsigned short* wp2  = (unsigned short*)take((size_t)131072 * 2);
    float* agg1 = buf;
    float* agg2 = buf;

    hipMemsetAsync(cnt,  0, (size_t)N_NODES * 4, stream);
    hipMemsetAsync(agg1, 0, (size_t)N_NODES * D_IN * 4, stream);
    hipMemsetAsync(out,  0, (size_t)out_size * 4, stream);

    // pack weights into WMMA B-fragment order (bf16)
    gnn_pack_weights<<<65536 / 256, 256, 0, stream>>>(W1l, W1r, D_IN, wp1, 65536);
    gnn_pack_weights<<<131072 / 256, 256, 0, stream>>>(W2l, W2r, HIDDEN, wp2, 131072);

    // degree + layer-1 mean aggregation
    gnn_count_edges<<<N_EDGES / 256, 256, 0, stream>>>(dst, cnt);
    gnn_edge_agg_f32<<<N_EDGES / 8, 256, 0, stream>>>(x, src, dst, agg1);

    // h = relu([mean|x] @ [W1_l;W1_r] + b1) -> bf16
    gnn_sage_gemm<D_IN, D_IN, false, false><<<N_NODES / 16, 512, 0, stream>>>(
        agg1, cnt, x, (const unsigned short*)nullptr, wp1, b1, hbf, nullptr, nullptr);

    // layer-2 aggregation (agg2 aliases agg1; zero it after GEMM1 in stream order)
    hipMemsetAsync(agg2, 0, (size_t)N_NODES * HIDDEN * 4, stream);
    gnn_edge_agg_bf16<<<N_EDGES / 8, 256, 0, stream>>>(hbf, src, dst, agg2);

    // out[g] += [mean2|h] @ [W2_l;W2_r] + b2 (fused global_add_pool)
    gnn_sage_gemm<HIDDEN, HIDDEN, true, true><<<N_NODES / 16, 512, 0, stream>>>(
        agg2, cnt, (const float*)nullptr, hbf, wp2, b2, nullptr, out, batch);
}
